// PAM_Module_v2_42872363549058
// MI455X (gfx1250) — matrix operand order repaired and checked by reading, not compiled, not run on hardware
//
#include <hip/hip_runtime.h>
#include <hip/hip_bf16.h>
#include <math.h>

typedef __bf16 bf16;
typedef __attribute__((ext_vector_type(16))) __bf16 v16bf;
typedef __attribute__((ext_vector_type(8)))  __bf16 v8bf;
typedef __attribute__((ext_vector_type(8)))  float  v8f;

#define Bn   4
#define Cn   256
#define C8n  32
#define Nn   4096
#define LOG2E 1.44269504088896340736f

__device__ __forceinline__ float fast_exp2(float x) {
  return __builtin_amdgcn_exp2f(x);   // raw v_exp_f32 (hardware exp2)
}

__device__ __forceinline__ v8bf ld8(const bf16* p) { return *(const v8bf*)p; }

__device__ __forceinline__ v16bf cat8(v8bf a, v8bf b) {
  v16bf r;
#pragma unroll
  for (int i = 0; i < 8; ++i) { r[i] = a[i]; r[i + 8] = b[i]; }
  return r;
}

// ---------------------------------------------------------------------------
// Kernel 1: 1x1-conv projections (f32 math, bf16 outputs laid out for WMMA)
//   Qf[b][n][32] = (Wk * x[:,n] + bk) * log2(e)   (flash "queries", exp2 domain)
//   Kf[b][n][32] =  Wq * x[:,n] + bq              (flash "keys")
//   Vt[b][c][n'] =  Wv * x[:,n] + bv              (flash "values", channel-major,
//                    n permuted inside each 64-chunk: n' = (n&15)*4 + ((n&63)>>4)
//                    to match the packed P-tile K order in the flash kernel)
// ---------------------------------------------------------------------------
#define PT 32  // pixels per block

__global__ __launch_bounds__(256)
void pam_proj(const float* __restrict__ x,
              const float* __restrict__ Wq, const float* __restrict__ bq,
              const float* __restrict__ Wk, const float* __restrict__ bk,
              const float* __restrict__ Wv, const float* __restrict__ bv,
              bf16* __restrict__ Qf, bf16* __restrict__ Kf, bf16* __restrict__ Vt)
{
  __shared__ __align__(16) float xs[Cn][PT + 4];
  const int tid = threadIdx.x;
  const int b   = blockIdx.x / (Nn / PT);
  const int n0  = (blockIdx.x % (Nn / PT)) * PT;

  const float* xb = x + (size_t)b * Cn * Nn;
  for (int idx = tid; idx < Cn * PT; idx += 256) {
    int c = idx >> 5, n = idx & 31;
    xs[c][n] = xb[(size_t)c * Nn + n0 + n];
  }
  __syncthreads();

  // ---- q / k projections: thread -> output channel o = tid%32, 4 pixels ----
  {
    const int o  = tid & 31;
    const int nb = tid >> 5;              // 0..7
#pragma unroll
    for (int j = 0; j < 4; ++j) {
      const int n = nb + 8 * j;           // 0..31
      float aq = bq[o], ak = bk[o];
#pragma unroll 8
      for (int c = 0; c < Cn; ++c) {
        const float xv = xs[c][n];
        aq = fmaf(Wq[o * Cn + c], xv, aq);
        ak = fmaf(Wk[o * Cn + c], xv, ak);
      }
      const size_t row = ((size_t)b * Nn + n0 + n) * C8n;
      Kf[row + o] = (bf16)aq;             // flash keys    = reference q rows
      Qf[row + o] = (bf16)(ak * LOG2E);   // flash queries = reference k cols
    }
  }

  // ---- v projection: thread owns channel o = tid, all 32 pixels ----
  {
    const int o = tid;
    float acc[PT];
#pragma unroll
    for (int n = 0; n < PT; ++n) acc[n] = bv[o];
    for (int c = 0; c < Cn; ++c) {
      const float w = Wv[o * Cn + c];
      const float4* xrow = (const float4*)&xs[c][0];
#pragma unroll
      for (int nq = 0; nq < PT / 4; ++nq) {
        const float4 xv = xrow[nq];
        acc[nq * 4 + 0] = fmaf(w, xv.x, acc[nq * 4 + 0]);
        acc[nq * 4 + 1] = fmaf(w, xv.y, acc[nq * 4 + 1]);
        acc[nq * 4 + 2] = fmaf(w, xv.z, acc[nq * 4 + 2]);
        acc[nq * 4 + 3] = fmaf(w, xv.w, acc[nq * 4 + 3]);
      }
    }
    // permuted store: within each 64-chunk, n -> (n&15)*4 + (n_local>>4)
    const int chunk = n0 & ~63;
    bf16* vbase = Vt + ((size_t)b * Cn + o) * Nn + chunk;
#pragma unroll
    for (int n = 0; n < PT; ++n) {
      const int nl = (n0 & 32) + n;                 // 0..63 within chunk
      const int ks = (nl & 15) * 4 + (nl >> 4);     // packed K order
      vbase[ks] = (bf16)acc[n];
    }
  }
}

// ---------------------------------------------------------------------------
// Kernel 2: flash attention over the swapped problem (softmax over axis=1).
//   out[b][c][m] = gamma * (sum_n softmax_n(S[m][n]) * Vt[c][n]) + x[b][c][m]
// n-step = 64: 4 score WMMAs, 8 accumulate WMMAs, 2 row-sum WMMAs (P @ ones
// broadcasts each row sum to every column, landing in the C/D lane layout).
// P tile is stored to LDS in a permuted-K order so each lane emits one packed
// b64 store; Vt was written with the identical K permutation.
// ---------------------------------------------------------------------------
#define NT 64

__global__ __launch_bounds__(128)
void pam_flash(const float* __restrict__ x,
               const bf16* __restrict__ Qf, const bf16* __restrict__ Kf,
               const bf16* __restrict__ Vt, const float* __restrict__ gamma_p,
               float* __restrict__ out)
{
  __shared__ __align__(16) bf16 Plds[4][16][NT];   // wave-private P tiles (8KB)

  const int tid  = threadIdx.x;
  const int wave = tid >> 5;
  const int lane = tid & 31;
  const int lo16 = (lane < 16);
  const int lr   = lane & 15;
  const int b    = blockIdx.x >> 8;
  const int m0   = (blockIdx.x & 255) << 4;
  const int c0   = wave * 64;
  const float gamma = gamma_p[0];

  // A operand: Qf rows m0..m0+15 (16x32 bf16), constant across the n loop.
  const bf16* qrow = Qf + ((size_t)b * Nn + m0 + lr) * C8n + (lo16 ? 0 : 8);
  const v16bf aQ = cat8(ld8(qrow), ld8(qrow + 16));

  // all-ones B operand: P @ ones -> every column holds the row sum
  v16bf ones;
#pragma unroll
  for (int i = 0; i < 16; ++i) ones[i] = (bf16)1.0f;

  v8f O[4], L;
  float mrun[8], scv[8];
#pragma unroll
  for (int g = 0; g < 4; ++g)
#pragma unroll
    for (int r = 0; r < 8; ++r) O[g][r] = 0.f;
#pragma unroll
  for (int r = 0; r < 8; ++r) { mrun[r] = -INFINITY; L[r] = 0.f; }

  const bf16* Kb = Kf + (size_t)b * Nn * C8n;
  const bf16* Vb = Vt + (size_t)b * Cn * Nn;

  for (int ni = 0; ni < Nn; ni += NT) {
    if (ni + NT < Nn) {  // pull next tiles toward L0 (global_prefetch_b8)
      __builtin_prefetch(Kb + (size_t)(ni + NT + lane) * C8n, 0, 0);
      __builtin_prefetch(Vb + (size_t)(c0 + lane) * Nn + ni + NT, 0, 0);
    }

    // ---- batched K loads (one clause), then 4 score WMMAs ----
    v16bf bK[4];
#pragma unroll
    for (int s = 0; s < 4; ++s) {
      const bf16* krow = Kb + (size_t)(ni + s * 16 + lr) * C8n + (lo16 ? 0 : 8);
      bK[s] = cat8(ld8(krow), ld8(krow + 16));
    }
    v8f S[4];
#pragma unroll
    for (int s = 0; s < 4; ++s) {
      v8f z;
#pragma unroll
      for (int r = 0; r < 8; ++r) z[r] = 0.f;
      S[s] = __builtin_amdgcn_wmma_f32_16x16x32_bf16(
          false, aQ, false, bK[s], (short)0, z, false, false);
    }

    // ---- online softmax (exp2 domain); row max via 4-stage butterfly ----
    const int q4 = lr * 4;   // packed K slot for this lane's 4 columns
#pragma unroll
    for (int r = 0; r < 8; ++r) {
      const float t0 = S[0][r], t1 = S[1][r], t2 = S[2][r], t3 = S[3][r];
      float tmax = fmaxf(fmaxf(t0, t1), fmaxf(t2, t3));
#pragma unroll
      for (int msk = 8; msk >= 1; msk >>= 1)
        tmax = fmaxf(tmax, __shfl_xor(tmax, msk, 32));
      const float mnew = fmaxf(mrun[r], tmax);
      scv[r]  = fast_exp2(mrun[r] - mnew);
      mrun[r] = mnew;
      const int ml = lo16 ? r : (r + 8);
      union { bf16 h[4]; uint2 u; } pk;
      pk.h[0] = (bf16)fast_exp2(t0 - mnew);
      pk.h[1] = (bf16)fast_exp2(t1 - mnew);
      pk.h[2] = (bf16)fast_exp2(t2 - mnew);
      pk.h[3] = (bf16)fast_exp2(t3 - mnew);
      *(uint2*)&Plds[wave][ml][q4] = pk.u;   // one b64 store per row
    }

    // reload P as two A operands (16m x 32k each); wave-private, DS in-order
    const bf16* prow = &Plds[wave][lr][lo16 ? 0 : 8];
    const v16bf aP0 = cat8(ld8(prow),      ld8(prow + 16));
    const v16bf aP1 = cat8(ld8(prow + 32), ld8(prow + 48));

    // ---- running denominator: L = L*sc + rowsum(P) via P @ ones ----
#pragma unroll
    for (int r = 0; r < 8; ++r) L[r] *= scv[r];
    L = __builtin_amdgcn_wmma_f32_16x16x32_bf16(false, aP0, false, ones,
                                                (short)0, L, false, false);
    L = __builtin_amdgcn_wmma_f32_16x16x32_bf16(false, aP1, false, ones,
                                                (short)0, L, false, false);

    // ---- accumulate O += P @ V for this wave's 4 channel groups ----
#pragma unroll
    for (int g = 0; g < 4; ++g) {
#pragma unroll
      for (int r = 0; r < 8; ++r) O[g][r] *= scv[r];
      const bf16* vrow = Vb + (size_t)(c0 + g * 16 + lr) * Nn + ni + (lo16 ? 0 : 8);
      const v16bf bV0 = cat8(ld8(vrow),      ld8(vrow + 16));
      const v16bf bV1 = cat8(ld8(vrow + 32), ld8(vrow + 48));
      O[g] = __builtin_amdgcn_wmma_f32_16x16x32_bf16(false, aP0, false, bV0,
                                                     (short)0, O[g], false, false);
      O[g] = __builtin_amdgcn_wmma_f32_16x16x32_bf16(false, aP1, false, bV1,
                                                     (short)0, O[g], false, false);
    }
  }

  // ---- normalize, residual, scatter to out[b][c][m] ----
  const float* xb = x   + (size_t)b * Cn * Nn;
  float*       ob = out + (size_t)b * Cn * Nn;
#pragma unroll
  for (int g = 0; g < 4; ++g) {
#pragma unroll
    for (int r = 0; r < 8; ++r) {
      const int mg = m0 + (lo16 ? r : r + 8);
      const int c  = c0 + g * 16 + lr;
      const size_t idx = (size_t)c * Nn + mg;
      ob[idx] = fmaf(gamma, O[g][r] / L[r], xb[idx]);
    }
  }
}

// ---------------------------------------------------------------------------
extern "C" void kernel_launch(void* const* d_in, const int* in_sizes, int n_in,
                              void* d_out, int out_size, void* d_ws, size_t ws_size,
                              hipStream_t stream) {
  (void)in_sizes; (void)n_in; (void)out_size; (void)ws_size;
  const float* x     = (const float*)d_in[0];
  const float* Wq    = (const float*)d_in[1];
  const float* bq    = (const float*)d_in[2];
  const float* Wk    = (const float*)d_in[3];
  const float* bk    = (const float*)d_in[4];
  const float* Wv    = (const float*)d_in[5];
  const float* bv    = (const float*)d_in[6];
  const float* gamma = (const float*)d_in[7];
  float* out = (float*)d_out;

  // workspace: Qf | Kf | Vt  (bf16)  = 1MB + 1MB + 8MB
  bf16* Qf = (bf16*)d_ws;
  bf16* Kf = Qf + (size_t)Bn * Nn * C8n;
  bf16* Vt = Kf + (size_t)Bn * Nn * C8n;

  pam_proj<<<Bn * (Nn / PT), 256, 0, stream>>>(x, Wq, bq, Wk, bk, Wv, bv, Qf, Kf, Vt);
  pam_flash<<<Bn * (Nn / 16), 128, 0, stream>>>(x, Qf, Kf, Vt, gamma, out);
}
